// Net_15324443312383
// MI455X (gfx1250) — matrix operand order repaired and checked by reading, not compiled, not run on hardware
//
#include <hip/hip_runtime.h>
#include <hip/hip_bf16.h>
#include <stdint.h>

// ---------------------------------------------------------------------------
// GCN pipeline for MI455X (gfx1250, wave32).
// Dense layers: v_wmma_f32_16x16x32_bf16, operands pre-packed bf16 so the
// inner loop is pure b128 loads + WMMA.
// Edge aggregation: CSR-by-dst gather (built once, reused by both convs);
// zero f32 atomics, with self-loop + bias + ReLU fused into the gather.
// ---------------------------------------------------------------------------

typedef __attribute__((ext_vector_type(16))) __bf16 v16bf;
typedef __attribute__((ext_vector_type(8)))  float  v8f;

#define WORDF   300
#define RGBF    512
#define XIN     812     // WORDF + RGBF
#define KPAD_IN 832     // XIN padded to multiple of 32
#define MIDF    256
#define ATTRF   64
#define OUTF    256
#define EMBF    512
#define KCAT    320     // MIDF + ATTRF

__device__ __forceinline__ unsigned short f2bf(float x) {
    unsigned int u = __float_as_uint(x);
    u = u + 0x7FFFu + ((u >> 16) & 1u);        // round-to-nearest-even
    return (unsigned short)(u >> 16);
}

union Frag { v16bf v; uint4 q[2]; };

__device__ __forceinline__ void atomAddF(float* p, float v) {
    __hip_atomic_fetch_add(p, v, __ATOMIC_RELAXED, __HIP_MEMORY_SCOPE_AGENT);
}

// ---------------- operand packing -----------------------------------------

// x [N,812] f32 -> xbf [N,832] bf16 zero-padded.
__global__ void k_cvt_x(const float* __restrict__ x,
                        unsigned short* __restrict__ xbf, int total) {
    int i = blockIdx.x * blockDim.x + threadIdx.x;
    if (i >= total) return;
    int n = i / KPAD_IN, k = i - n * KPAD_IN;
    xbf[i] = (k < XIN) ? f2bf(x[(size_t)n * XIN + k]) : (unsigned short)0;
}

// attributes [N,64] f32 -> bf16 columns 256..319 of h1cat [N,320].
__global__ void k_pack_attr(const float* __restrict__ attr,
                            unsigned short* __restrict__ h1cat, int total) {
    int i = blockIdx.x * blockDim.x + threadIdx.x;
    if (i >= total) return;
    int n = i >> 6, j = i & 63;
    h1cat[(size_t)n * KCAT + MIDF + j] = f2bf(attr[i]);
}

// Input layer weights: block-diagonal [832 x 256], transposed bf16 Wt[c][k].
__global__ void k_pack_win(const float* __restrict__ Ww,
                           const float* __restrict__ Wr,
                           unsigned short* __restrict__ Wt) {
    int idx = blockIdx.x * blockDim.x + threadIdx.x;   // c*KPAD_IN + k
    if (idx >= 256 * KPAD_IN) return;
    int c = idx / KPAD_IN, k = idx - c * KPAD_IN;
    float v = 0.f;
    if (c < 128) { if (k < WORDF) v = Ww[k * 128 + c]; }
    else         { if (k >= WORDF && k < XIN) v = Wr[(k - WORDF) * 128 + (c - 128)]; }
    Wt[idx] = f2bf(v);
}

// Generic: W [K x 256] row-major f32 -> Wt[c*K + k] bf16 (transposed).
__global__ void k_pack_wt(const float* __restrict__ W,
                          unsigned short* __restrict__ Wt, int K) {
    int idx = blockIdx.x * blockDim.x + threadIdx.x;
    if (idx >= K * 256) return;
    int c = idx / K, k = idx - c * K;
    Wt[idx] = f2bf(W[k * 256 + c]);
}

// ---------------- WMMA GEMM: C[N,256] = A[N,Kpad](bf16) @ Wt(bf16) ---------
// Wave computes 16 rows x 64 cols (4 accumulators); block = 8 waves.
// Inner loop: 2 b128 A-loads + 8 b128 B-loads, then 4 WMMAs.
__global__ __launch_bounds__(256) void k_gemm_bf16(
    const unsigned short* __restrict__ A, int ka,      // A row stride (elems)
    int Kpad,
    const unsigned short* __restrict__ Wt, int kb,     // Wt col stride (elems)
    const float* __restrict__ bias0, const float* __restrict__ bias1, int bsplit,
    float* __restrict__ Cf,                            // f32 out (or null)
    unsigned short* __restrict__ Cbf, int ldcb,        // bf16 out (or null)
    int Nrows, int relu)
{
    const int lane  = threadIdx.x & 31;
    const int wave  = threadIdx.x >> 5;
    const int rowTile = blockIdx.x * 2 + (wave >> 2);
    const int colBase = (wave & 3) * 64;
    const int row0  = rowTile * 16;
    if (row0 >= Nrows) return;                 // wave-uniform: EXEC stays full

    const int l16   = lane & 15;
    const int khalf = lane >> 4;
    const int m     = row0 + l16;

    // Per-lane base pointers (ISA fragment layouts), indexed by kk.
    const unsigned short* aBase = A + (size_t)m * ka + khalf * 8;
    const unsigned short* bBase[4];
#pragma unroll
    for (int s = 0; s < 4; ++s) {
        const int c = colBase + s * 16 + l16;
        bBase[s] = Wt + (size_t)c * kb + khalf * 8;
    }

    const v8f vzero = {0.f, 0.f, 0.f, 0.f, 0.f, 0.f, 0.f, 0.f};
    v8f acc[4];
#pragma unroll
    for (int s = 0; s < 4; ++s) acc[s] = vzero;

    for (int kk = 0; kk < Kpad; kk += 32) {
        Frag a;
        const uint4* pa = (const uint4*)(aBase + kk);
        a.q[0] = pa[0];                        // K = kk+khalf*8 .. +7
        a.q[1] = pa[2];                        // K = kk+16+khalf*8 .. +7
        Frag b[4];
#pragma unroll
        for (int s = 0; s < 4; ++s) {
            const uint4* pb = (const uint4*)(bBase[s] + kk);
            b[s].q[0] = pb[0];
            b[s].q[1] = pb[2];
        }
#pragma unroll
        for (int s = 0; s < 4; ++s)
            acc[s] = __builtin_amdgcn_wmma_f32_16x16x32_bf16(
                false, a.v, false, b[s].v, (short)0, acc[s], false, false);
    }

    // Epilogue: D layout -> VGPR v holds row v (lanes 0-15) / v+8 (16-31).
#pragma unroll
    for (int s = 0; s < 4; ++s) {
#pragma unroll
        for (int v = 0; v < 8; ++v) {
            const int mr = row0 + v + khalf * 8;
            const int c  = colBase + s * 16 + l16;
            float bv = 0.f;
            if (bias0) bv = (bias1 && c >= bsplit) ? bias1[c - bsplit] : bias0[c];
            float val = acc[s][v] + bv;
            if (relu) val = fmaxf(val, 0.f);
            if (mr < Nrows) {
                if (Cf)  Cf[(size_t)mr * 256 + c] = val;
                if (Cbf) Cbf[(size_t)mr * ldcb + c] = f2bf(val);
            }
        }
    }
}

// ---------------- degree / normalization ----------------------------------

__global__ void k_fill(float* __restrict__ p, float v, int n) {
    int i = blockIdx.x * blockDim.x + threadIdx.x;
    if (i < n) p[i] = v;
}

__global__ void k_fill_i32(int* __restrict__ p, int v, int n) {
    int i = blockIdx.x * blockDim.x + threadIdx.x;
    if (i < n) p[i] = v;
}

__global__ void k_deg_edges(const int* __restrict__ ei,
                            const float* __restrict__ ew,
                            float* __restrict__ deg, int E) {
    int e = blockIdx.x * blockDim.x + threadIdx.x;
    if (e >= E) return;
    atomAddF(&deg[ei[E + e]], ew[e]);          // dst row of edge_index
}

__global__ void k_rsqrt(float* __restrict__ p, int n) {
    int i = blockIdx.x * blockDim.x + threadIdx.x;
    if (i < n) p[i] = rsqrtf(p[i]);
}

// ---------------- CSR build (by destination) -------------------------------

__global__ void k_count(const int* __restrict__ ei, int* __restrict__ cnt, int E) {
    int e = blockIdx.x * blockDim.x + threadIdx.x;
    if (e >= E) return;
    atomicAdd(&cnt[ei[E + e]], 1);
}

// Single-block exclusive scan: off[0..N] (off[N]=E), ptr = copy of off.
__global__ __launch_bounds__(1024) void k_scan(
    const int* __restrict__ cnt, int* __restrict__ off,
    int* __restrict__ ptr, int N)
{
    __shared__ int sdata[1024];
    __shared__ int carry;
    const int tid = threadIdx.x;
    if (tid == 0) carry = 0;
    __syncthreads();
    for (int base = 0; base < N; base += 1024) {
        const int i = base + tid;
        const int v = (i < N) ? cnt[i] : 0;
        sdata[tid] = v;
        __syncthreads();
#pragma unroll
        for (int d = 1; d < 1024; d <<= 1) {    // Hillis-Steele inclusive scan
            int t = (tid >= d) ? sdata[tid - d] : 0;
            __syncthreads();
            sdata[tid] += t;
            __syncthreads();
        }
        const int excl = carry + sdata[tid] - v;
        if (i < N) { off[i] = excl; ptr[i] = excl; }
        __syncthreads();
        if (tid == 0) carry += sdata[1023];
        __syncthreads();
    }
    if (tid == 0) off[N] = carry;
}

// Scatter (src, norm) into dst-sorted CSR segments via ticket atomics.
__global__ void k_fill_csr(const int* __restrict__ ei,
                           const float* __restrict__ ew,
                           const float* __restrict__ dinv,
                           int* __restrict__ ptr,
                           int* __restrict__ srcS,
                           float* __restrict__ normS, int E)
{
    int e = blockIdx.x * blockDim.x + threadIdx.x;
    if (e >= E) return;
    const int s = ei[e], d = ei[E + e];
    const int p = atomicAdd(&ptr[d], 1);
    srcS[p] = s;
    normS[p] = dinv[s] * ew[e] * dinv[d];
}

// ---------------- fused GCN aggregation (gather) ---------------------------
// One wave per dst node: out[d] = act(sum_e norm*xw[src] + dinv^2*xw[d] + b).
// ~16 edges/node; row gathers hit L2 (xw is ~20 MB). No f32 atomics.
__global__ __launch_bounds__(256) void k_gcn_agg(
    const float* __restrict__ xw,
    const int* __restrict__ srcS, const float* __restrict__ normS,
    const int* __restrict__ off, const float* __restrict__ dinv,
    const float* __restrict__ bias,
    float* __restrict__ outF,                   // f32 out (or null)
    unsigned short* __restrict__ outBf, int ldbf, // bf16 out (or null)
    int Nn)
{
    const int gt = blockIdx.x * blockDim.x + threadIdx.x;
    const int d = gt >> 5, lane = gt & 31;
    if (d >= Nn) return;

    const float di = dinv[d];
    const float selfw = di * di;
    const float* xd = xw + (size_t)d * 256;

    float acc[8];
#pragma unroll
    for (int j = 0; j < 8; ++j) {
        const int f = j * 32 + lane;
        acc[j] = fmaf(selfw, xd[f], bias[f]);
    }

    const int beg = off[d], end = off[d + 1];
    for (int i = beg; i < end; ++i) {
        const int s = srcS[i];                 // wave-uniform
        const float nw = normS[i];             // wave-uniform
        const float* xs = xw + (size_t)s * 256;
#pragma unroll
        for (int j = 0; j < 8; ++j)
            acc[j] = fmaf(nw, xs[j * 32 + lane], acc[j]);
    }

#pragma unroll
    for (int j = 0; j < 8; ++j) {
        const int f = j * 32 + lane;
        const float v = fmaxf(acc[j], 0.f);    // both convs apply ReLU
        if (outF)  outF[(size_t)d * 256 + f] = v;
        if (outBf) outBf[(size_t)d * ldbf + f] = f2bf(v);
    }
}

// ---------------- mean-pool column sums over [h2 || attr] ------------------
__global__ __launch_bounds__(320) void k_colsum(
    const float* __restrict__ h2, const float* __restrict__ attr,
    float* __restrict__ colsum, int Nn)
{
    const int f = threadIdx.x;                 // 0..319
    const int per = (Nn + gridDim.x - 1) / gridDim.x;
    int start = blockIdx.x * per;
    int end = start + per; if (end > Nn) end = Nn;
    float s = 0.f;
    if (f < 256) { for (int n = start; n < end; ++n) s += h2[(size_t)n * 256 + f]; }
    else { const int g = f - 256;
           for (int n = start; n < end; ++n) s += attr[(size_t)n * 64 + g]; }
    atomAddF(&colsum[f], s);
}

// out[j] = relu(b_map[j] + sum_k mean[k] * W_map[k,j])
__global__ void k_final(const float* __restrict__ colsum,
                        const float* __restrict__ Wm,
                        const float* __restrict__ bm,
                        float* __restrict__ out, int Nn) {
    int j = blockIdx.x * blockDim.x + threadIdx.x;
    if (j >= EMBF) return;
    const float inv = 1.0f / (float)Nn;
    float acc = bm[j];
    for (int k = 0; k < OUTF + ATTRF; ++k)
        acc = fmaf(colsum[k] * inv, Wm[(size_t)k * EMBF + j], acc);
    out[j] = fmaxf(acc, 0.f);
}

// ---------------------------------------------------------------------------

extern "C" void kernel_launch(void* const* d_in, const int* in_sizes, int n_in,
                              void* d_out, int out_size, void* d_ws, size_t ws_size,
                              hipStream_t stream) {
    const float* x      = (const float*)d_in[0];
    const float* attrs  = (const float*)d_in[1];
    const float* ew     = (const float*)d_in[2];
    const float* W_word = (const float*)d_in[3];
    const float* b_word = (const float*)d_in[4];
    const float* W_rgb  = (const float*)d_in[5];
    const float* b_rgb  = (const float*)d_in[6];
    const float* W1     = (const float*)d_in[7];
    const float* b1     = (const float*)d_in[8];
    const float* W2     = (const float*)d_in[9];
    const float* b2     = (const float*)d_in[10];
    const float* W_map  = (const float*)d_in[11];
    const float* b_map  = (const float*)d_in[12];
    const int*   ei     = (const int*)d_in[13];

    const int E = in_sizes[2];
    const int N = in_sizes[1] / ATTRF;
    float* out = (float*)d_out;

    // ---- workspace carve (256B aligned slices) ----
    char* wp = (char*)d_ws;
    auto carve = [&](size_t bytes) -> void* {
        void* p = (void*)wp;
        wp += (bytes + 255) & ~(size_t)255;
        return p;
    };
    float* dinv  = (float*)carve((size_t)N * 4);
    unsigned short* xbf   = (unsigned short*)carve((size_t)N * KPAD_IN * 2);
    unsigned short* h0bf  = (unsigned short*)carve((size_t)N * 256 * 2);
    unsigned short* h1cat = (unsigned short*)carve((size_t)N * KCAT * 2);
    float* xw    = (float*)carve((size_t)N * 256 * 4);
    float* h2    = (float*)carve((size_t)N * 256 * 4);
    unsigned short* Win = (unsigned short*)carve((size_t)256 * KPAD_IN * 2);
    unsigned short* W1t = (unsigned short*)carve((size_t)256 * 256 * 2);
    unsigned short* W2t = (unsigned short*)carve((size_t)256 * 320 * 2);
    int*   cnt   = (int*)carve((size_t)N * 4);
    int*   off   = (int*)carve((size_t)(N + 1) * 4);
    int*   ptr   = (int*)carve((size_t)N * 4);
    int*   srcS  = (int*)carve((size_t)E * 4);
    float* normS = (float*)carve((size_t)E * 4);
    float* colsum = (float*)carve(320 * 4);

    const int rowTiles = (N + 15) / 16;
    const int gemmBlocks = (rowTiles + 1) / 2;
    const int aggBlocks = (N * 32 + 255) / 256;

    // ---- pack operands to bf16 ----
    k_cvt_x<<<((N * KPAD_IN) + 255) / 256, 256, 0, stream>>>(x, xbf, N * KPAD_IN);
    k_pack_attr<<<((N * ATTRF) + 255) / 256, 256, 0, stream>>>(attrs, h1cat, N * ATTRF);
    k_pack_win<<<(256 * KPAD_IN + 255) / 256, 256, 0, stream>>>(W_word, W_rgb, Win);
    k_pack_wt <<<(256 * 256 + 255) / 256, 256, 0, stream>>>(W1, W1t, 256);
    k_pack_wt <<<(256 * 320 + 255) / 256, 256, 0, stream>>>(W2, W2t, 320);

    // ---- degree -> dinv = rsqrt(1 + sum_in ew) ----
    k_fill<<<(N + 255) / 256, 256, 0, stream>>>(dinv, 1.0f, N);
    k_deg_edges<<<(E + 255) / 256, 256, 0, stream>>>(ei, ew, dinv, E);
    k_rsqrt<<<(N + 255) / 256, 256, 0, stream>>>(dinv, N);

    // ---- CSR by destination (built once, reused by both convs) ----
    k_fill_i32<<<(N + 255) / 256, 256, 0, stream>>>(cnt, 0, N);
    k_count<<<(E + 255) / 256, 256, 0, stream>>>(ei, cnt, E);
    k_scan<<<1, 1024, 0, stream>>>(cnt, off, ptr, N);
    k_fill_csr<<<(E + 255) / 256, 256, 0, stream>>>(ei, ew, dinv, ptr, srcS, normS, E);

    // ---- layer 0: h0bf = bf16(relu(x @ Win + [b_word||b_rgb])) ----
    k_gemm_bf16<<<gemmBlocks, 256, 0, stream>>>(
        xbf, KPAD_IN, KPAD_IN, Win, KPAD_IN,
        b_word, b_rgb, 128,
        (float*)nullptr, h0bf, 256, N, 1);

    // ---- GCNConv1: xw = h0 @ W1; gather-agg -> h1cat[:,0:256] (bf16) ----
    k_gemm_bf16<<<gemmBlocks, 256, 0, stream>>>(
        h0bf, 256, 256, W1t, 256,
        (const float*)nullptr, (const float*)nullptr, 0,
        xw, (unsigned short*)nullptr, 0, N, 0);
    k_gcn_agg<<<aggBlocks, 256, 0, stream>>>(
        xw, srcS, normS, off, dinv, b1,
        (float*)nullptr, h1cat, KCAT, N);

    // ---- GCNConv2 on h1cat [N,320]: xw = h1cat @ W2; gather-agg -> h2 ----
    k_gemm_bf16<<<gemmBlocks, 256, 0, stream>>>(
        h1cat, KCAT, KCAT, W2t, KCAT,
        (const float*)nullptr, (const float*)nullptr, 0,
        xw, (unsigned short*)nullptr, 0, N, 0);
    k_gcn_agg<<<aggBlocks, 256, 0, stream>>>(
        xw, srcS, normS, off, dinv, b2,
        h2, (unsigned short*)nullptr, 0, N);

    // ---- mean-pool over [h2 || attr], final linear + relu ----
    k_fill<<<2, 256, 0, stream>>>(colsum, 0.f, 320);
    k_colsum<<<80, 320, 0, stream>>>(h2, attrs, colsum, N);
    k_final<<<(EMBF + 255) / 256, 256, 0, stream>>>(colsum, W_map, b_map, out, N);
}
